// GenesetEncoder_5420248728198
// MI455X (gfx1250) — compile-verified, matrix operand order checked by reading, not run on hardware
//
#include <hip/hip_runtime.h>

typedef __attribute__((ext_vector_type(16))) __bf16 v16bf;
typedef __attribute__((ext_vector_type(8)))  __bf16 v8bf;
typedef __attribute__((ext_vector_type(8)))  float  v8f;

#define BATCH   256
#define NSETS   1000
#define NGENES  10000
#define G       128
#define H1      64
#define H2      32
#define EPSV    1e-5f

// element strides (padded to dodge bank conflicts, keep 16B alignment)
#define XG_STR  144   // bf16 [256][G]
#define WT1_STR 136   // bf16 [H1][G]   (W1 transposed)
#define H1F_STR 68    // f32  [256][H1]
#define H1B_STR 72    // bf16 [256][H1] (normalized, aliases xg region)
#define H2F_STR 36    // f32  [256][H2] (aliases xg region, offset 36864)
#define WT2_STR 72    // bf16 [H2][H1]  (W2 transposed, aliases Wt1 region)

// byte offsets inside dynamic LDS
#define OFF_R1   0        // 73728 B : xg -> (h1b @0, h2f @36864)
#define OFF_WT   73728    // 17408 B : Wt1 / Wt2
#define OFF_H1F  91136    // 69632 B : h1 f32
#define OFF_PART 160768   // 2048  B : 256 x float2 partial stats
#define OFF_SCL  162816   // 256   B : scale[64]
#define OFF_SHF  163072   // 256   B : shift[64]
#define OFF_IDX  163328   // 512   B : gene_idx cache
#define SMEM_BYTES 163840

__global__ __launch_bounds__(256) void geneset_encoder_kernel(
    const float* __restrict__ x,  const int* __restrict__ gidx,
    const float* __restrict__ W1, const float* __restrict__ b1,
    const float* __restrict__ g1, const float* __restrict__ be1,
    const float* __restrict__ W2, const float* __restrict__ b2,
    const float* __restrict__ g2, const float* __restrict__ be2,
    const float* __restrict__ W3, const float* __restrict__ b3,
    float* __restrict__ out)
{
    extern __shared__ char smem[];
    __bf16* xg    = (__bf16*)(smem + OFF_R1);
    __bf16* h1b   = (__bf16*)(smem + OFF_R1);          // alias (xg dead)
    float*  h2f   = (float*)(smem + OFF_R1 + 36864);   // alias (xg dead)
    __bf16* wt    = (__bf16*)(smem + OFF_WT);          // Wt1 then Wt2
    float*  h1f   = (float*)(smem + OFF_H1F);
    float2* part  = (float2*)(smem + OFF_PART);
    float*  scale = (float*)(smem + OFF_SCL);
    float*  shift = (float*)(smem + OFF_SHF);
    int*    idx   = (int*)(smem + OFF_IDX);

    const int s    = blockIdx.x;
    const int tid  = threadIdx.x;
    const int lane = tid & 31;
    const int wave = tid >> 5;
    const int r16  = lane & 15;            // row/col within 16-wide tile
    const int kb   = (lane >> 4) << 3;     // K sub-base: 0 (lanes<16) or 8
    const int mhi  = (lane >> 4) << 3;     // +8 row offset for lanes 16..31

    // ---- cache gene indices, gather x -> bf16 LDS, load W1^T -> bf16 LDS ----
    if (tid < G) idx[tid] = gidx[s * G + tid];
    __syncthreads();

    for (int i = tid; i < BATCH * G; i += 256) {
        int b = i >> 7, g = i & (G - 1);
        xg[b * XG_STR + g] = (__bf16)x[b * NGENES + idx[g]];
    }
    const float* W1s = W1 + (long long)s * G * H1;      // [G][H1]
    for (int i = tid; i < G * H1; i += 256) {
        int g = i >> 6, h = i & (H1 - 1);
        wt[h * WT1_STR + g] = (__bf16)W1s[i];
    }
    __syncthreads();

    // ---- layer 1: h1 = relu(xg @ W1 + b1), WMMA bf16 16x16x32 ----
    for (int tile = wave; tile < (BATCH / 16) * (H1 / 16); tile += 8) {
        int bt = tile >> 2, ht = tile & 3;
        v8f c = {};
        const __bf16* arow = xg + (bt * 16 + r16) * XG_STR;
        const __bf16* brow = wt + (ht * 16 + r16) * WT1_STR;
        for (int k0 = 0; k0 < G; k0 += 32) {
            v8bf alo = *(const v8bf*)(arow + k0 + kb);
            v8bf ahi = *(const v8bf*)(arow + k0 + kb + 16);
            v8bf blo = *(const v8bf*)(brow + k0 + kb);
            v8bf bhi = *(const v8bf*)(brow + k0 + kb + 16);
            v16bf a, bm;
#pragma unroll
            for (int e = 0; e < 8; ++e) {
                a[e] = alo[e];  a[8 + e] = ahi[e];
                bm[e] = blo[e]; bm[8 + e] = bhi[e];
            }
            c = __builtin_amdgcn_wmma_f32_16x16x32_bf16(
                    false, a, false, bm, (short)0, c, false, false);
        }
        float bias = b1[s * H1 + ht * 16 + r16];
#pragma unroll
        for (int r = 0; r < 8; ++r) {
            int m = bt * 16 + r + mhi;
            h1f[m * H1F_STR + ht * 16 + r16] = fmaxf(c[r] + bias, 0.f);
        }
    }
    __syncthreads();

    // ---- batchnorm 1 (stats over batch axis, exact) ----
    {
        int col = tid & 63, p = tid >> 6;   // 4 partials x 64 cols
        float sum = 0.f, sq = 0.f;
        for (int r = p * 64; r < p * 64 + 64; ++r) {
            float v = h1f[r * H1F_STR + col];
            sum += v; sq += v * v;
        }
        part[tid] = make_float2(sum, sq);
    }
    __syncthreads();
    if (tid < H1) {
        float sum = 0.f, sq = 0.f;
        for (int p = 0; p < 4; ++p) { float2 v = part[p * 64 + tid]; sum += v.x; sq += v.y; }
        float mean = sum * (1.f / 256.f);
        float var  = sq * (1.f / 256.f) - mean * mean;
        float inv  = rsqrtf(var + EPSV) * g1[s * H1 + tid];
        scale[tid] = inv;
        shift[tid] = be1[s * H1 + tid] - mean * inv;
    }
    __syncthreads();
    // normalize + cast to bf16 (into region aliasing dead xg)
    for (int i = tid; i < BATCH * H1; i += 256) {
        int r = i >> 6, c2 = i & 63;
        h1b[r * H1B_STR + c2] = (__bf16)(h1f[r * H1F_STR + c2] * scale[c2] + shift[c2]);
    }
    // load W2^T (overwrites Wt1, already consumed)
    const float* W2s = W2 + (long long)s * H1 * H2;      // [H1][H2]
    for (int i = tid; i < H1 * H2; i += 256) {
        int k = i >> 5, n = i & (H2 - 1);
        wt[n * WT2_STR + k] = (__bf16)W2s[i];
    }
    __syncthreads();

    // ---- layer 2: h2 = relu(h1n @ W2 + b2) ----
    for (int tile = wave; tile < (BATCH / 16) * (H2 / 16); tile += 8) {
        int bt = tile >> 1, ht = tile & 1;
        v8f c = {};
        const __bf16* arow = h1b + (bt * 16 + r16) * H1B_STR;
        const __bf16* brow = wt  + (ht * 16 + r16) * WT2_STR;
        for (int k0 = 0; k0 < H1; k0 += 32) {
            v8bf alo = *(const v8bf*)(arow + k0 + kb);
            v8bf ahi = *(const v8bf*)(arow + k0 + kb + 16);
            v8bf blo = *(const v8bf*)(brow + k0 + kb);
            v8bf bhi = *(const v8bf*)(brow + k0 + kb + 16);
            v16bf a, bm;
#pragma unroll
            for (int e = 0; e < 8; ++e) {
                a[e] = alo[e];  a[8 + e] = ahi[e];
                bm[e] = blo[e]; bm[8 + e] = bhi[e];
            }
            c = __builtin_amdgcn_wmma_f32_16x16x32_bf16(
                    false, a, false, bm, (short)0, c, false, false);
        }
        float bias = b2[s * H2 + ht * 16 + r16];
#pragma unroll
        for (int r = 0; r < 8; ++r) {
            int m = bt * 16 + r + mhi;
            h2f[m * H2F_STR + ht * 16 + r16] = fmaxf(c[r] + bias, 0.f);
        }
    }
    __syncthreads();

    // ---- batchnorm 2 stats ----
    {
        int col = tid & 31, p = tid >> 5;   // 8 partials x 32 cols
        float sum = 0.f, sq = 0.f;
        for (int r = p * 32; r < p * 32 + 32; ++r) {
            float v = h2f[r * H2F_STR + col];
            sum += v; sq += v * v;
        }
        part[tid] = make_float2(sum, sq);
    }
    __syncthreads();
    if (tid < H2) {
        float sum = 0.f, sq = 0.f;
        for (int p = 0; p < 8; ++p) { float2 v = part[p * 32 + tid]; sum += v.x; sq += v.y; }
        float mean = sum * (1.f / 256.f);
        float var  = sq * (1.f / 256.f) - mean * mean;
        float inv  = rsqrtf(var + EPSV) * g2[s * H2 + tid];
        scale[tid] = inv;
        shift[tid] = be2[s * H2 + tid] - mean * inv;
    }
    __syncthreads();

    // ---- layer 3: out[b,s] = relu(dot(bn(h2[b]), W3[s]) + b3[s]) ----
    {
        const float* W3s = W3 + s * H2;
        float acc = b3[s];
#pragma unroll
        for (int h = 0; h < H2; ++h)
            acc += (h2f[tid * H2F_STR + h] * scale[h] + shift[h]) * W3s[h];
        out[tid * NSETS + s] = fmaxf(acc, 0.f);
    }
}

extern "C" void kernel_launch(void* const* d_in, const int* in_sizes, int n_in,
                              void* d_out, int out_size, void* d_ws, size_t ws_size,
                              hipStream_t stream) {
    const float* x    = (const float*)d_in[0];
    const int*   gidx = (const int*)  d_in[1];
    const float* W1   = (const float*)d_in[2];
    const float* b1   = (const float*)d_in[3];
    const float* g1   = (const float*)d_in[4];
    const float* be1  = (const float*)d_in[5];
    const float* W2   = (const float*)d_in[6];
    const float* b2   = (const float*)d_in[7];
    const float* g2   = (const float*)d_in[8];
    const float* be2  = (const float*)d_in[9];
    const float* W3   = (const float*)d_in[10];
    const float* b3   = (const float*)d_in[11];
    float* out = (float*)d_out;

    (void)hipFuncSetAttribute((const void*)geneset_encoder_kernel,
                              hipFuncAttributeMaxDynamicSharedMemorySize,
                              SMEM_BYTES);
    geneset_encoder_kernel<<<NSETS, 256, SMEM_BYTES, stream>>>(
        x, gidx, W1, b1, g1, be1, W2, b2, g2, be2, W3, b3, out);
}
